// MultiHeadAttentionLayer_62251255988869
// MI455X (gfx1250) — matrix operand order repaired and checked by reading, not compiled, not run on hardware
//
#include <hip/hip_runtime.h>
#include <hip/hip_bf16.h>

// ---------------- constants (match reference) ----------------
#define BATCH   8
#define SEQ     1024
#define HIDDIM  1024
#define NHEADS  16
#define HDIM    64
#define TB      160   // padded relative-distance bins (129 used, rest zero)

typedef __bf16 bf16;
typedef __attribute__((ext_vector_type(16))) __bf16 v16bf;
typedef __attribute__((ext_vector_type(8)))  __bf16 v8bf;
typedef __attribute__((ext_vector_type(8)))  float  v8f;
typedef __attribute__((ext_vector_type(4)))  unsigned int v4u;
typedef __attribute__((ext_vector_type(4)))  int v4i;
typedef __attribute__((ext_vector_type(8)))  int v8i;

__device__ inline v8f vzero8() {
  v8f v;
  #pragma unroll
  for (int i = 0; i < 8; ++i) v[i] = 0.0f;
  return v;
}

__device__ inline v8f wmma_bf16(v16bf a, v16bf b, v8f c) {
  // D = A(16x32 bf16) * B(32x16 bf16) + C(16x16 f32)
  return __builtin_amdgcn_wmma_f32_16x16x32_bf16(
      /*neg_a=*/false, a, /*neg_b=*/false, b,
      /*c_mod=*/(short)0, c, /*reuse_a=*/false, /*reuse_b=*/false);
}

// A-operand loader: row-major [M][ld] bf16 source; per ISA 7.12.2 layout:
// lanes 0-15 (M=lane):  K = 0..7  then 16..23
// lanes 16-31(M=lane-16): K = 8..15 then 24..31
__device__ inline v16bf load_a_bf16(const bf16* A, int ld, int row0, int k0) {
  int lane = threadIdx.x & 31;
  int m  = lane & 15;
  int hi = lane >> 4;
  const bf16* p = A + (size_t)(row0 + m) * ld + k0 + hi * 8;
  union { v16bf v; v8bf h[2]; } u;
  u.h[0] = *(const v8bf*)(p);       // K chunk 0..7 / 8..15
  u.h[1] = *(const v8bf*)(p + 16);  // K chunk 16..23 / 24..31
  return u.v;
}

// B-operand loader from an [N][K]-row-major source (column-n strip is a
// contiguous K run). Per ISA: lanes 0-15 = N, K=0..7,16..23; lanes 16-31 = N, K=8..15,24..31.
__device__ inline v16bf load_b_bf16_nk(const bf16* Bt, int ld, int n0, int k0) {
  int lane = threadIdx.x & 31;
  int n  = lane & 15;
  int kh = lane >> 4;
  const bf16* p = Bt + (size_t)(n0 + n) * ld + k0 + kh * 8;
  union { v16bf v; v8bf h[2]; } u;
  u.h[0] = *(const v8bf*)(p);
  u.h[1] = *(const v8bf*)(p + 16);
  return u.v;
}

// B-operand loader from a [K][N]-row-major source (strided column reads; LDS).
__device__ inline v16bf load_b_bf16_kn(const bf16* Bkn, int ld, int k0, int n0) {
  int lane = threadIdx.x & 31;
  int n  = lane & 15;
  int kh = lane >> 4;
  const bf16* p = Bkn + (size_t)(k0 + kh * 8) * ld + n0 + n;
  v16bf v;
  #pragma unroll
  for (int t = 0; t < 16; ++t) v[t] = p[(size_t)(t + (t & 8)) * ld];
  return v;
}

// ---------------- Tensor Data Mover: 2D tile -> LDS ----------------
// D# per CDNA5 ISA: group0 = {count, lds_addr, global_addr[56:0], type=2},
// group1 = {data_size=2B, tensor_dim0/1, tile_dim0/1, tensor_dim0_stride}.
__device__ inline void tdm_load_2d(const bf16* lds_ptr, const bf16* gptr,
                                   unsigned td0, unsigned td1,
                                   unsigned t0, unsigned t1,
                                   unsigned long long stride0elems) {
  unsigned lds = (unsigned)(unsigned long long)(const void*)lds_ptr;
  unsigned long long ga = (unsigned long long)(const void*)gptr;
  v4u g0;
  g0[0] = 1u;                                                // count=1, user
  g0[1] = lds;                                               // lds_addr (bytes)
  g0[2] = (unsigned)ga;                                      // global_addr lo
  g0[3] = (unsigned)((ga >> 32) & 0x1FFFFFFull) | (2u << 30);// addr hi | type=2
  v8i g1;
  g1[0] = (int)(1u << 16);                                   // data_size=1 (2B)
  g1[1] = (int)((td0 & 0xFFFFu) << 16);                      // tensor_dim0 lo
  g1[2] = (int)(((td0 >> 16) & 0xFFFFu) | ((td1 & 0xFFFFu) << 16));
  g1[3] = (int)(((td1 >> 16) & 0xFFFFu) | ((t0 & 0xFFFFu) << 16)); // tile_dim0
  g1[4] = (int)(t1 & 0xFFFFu);                               // tile_dim1 (d2=0)
  g1[5] = (int)(stride0elems & 0xFFFFFFFFull);               // dim0 stride lo
  g1[6] = (int)((stride0elems >> 32) & 0xFFFFull);           // dim0 stride hi
  g1[7] = 0;
  v4i z4;
  #pragma unroll
  for (int i = 0; i < 4; ++i) z4[i] = 0;
#if defined(__clang_major__) && (__clang_major__ >= 23)
  v8i z8;
  #pragma unroll
  for (int i = 0; i < 8; ++i) z8[i] = 0;
  __builtin_amdgcn_tensor_load_to_lds(g0, g1, z4, z4, z8, 0);
#else
  __builtin_amdgcn_tensor_load_to_lds(g0, g1, z4, z4, 0);
#endif
}

// ---------------- fp32 -> bf16 conversion ----------------
__global__ void f32_to_bf16_kernel(const float* __restrict__ src,
                                   bf16* __restrict__ dst, int n) {
  int i = blockIdx.x * blockDim.x + threadIdx.x;
  if (i < n) dst[i] = (bf16)src[i];
}

// ---------------- fused QKV projection: out = X @ W^T + b ----------------
// Q,K stored [B,NH,S,HD]; V stored transposed [B,NH,HD,S] for the PV WMMA.
__global__ __launch_bounds__(256)
void qkv_gemm_kernel(const bf16* __restrict__ X,
                     const bf16* __restrict__ Wq, const bf16* __restrict__ Wk,
                     const bf16* __restrict__ Wv,
                     const float* __restrict__ bq, const float* __restrict__ bk,
                     const float* __restrict__ bv,
                     bf16* __restrict__ Qb, bf16* __restrict__ Kb,
                     bf16* __restrict__ Vtb) {
  int blk = blockIdx.x;
  int mat = blk / (64 * 16);
  int rem = blk % (64 * 16);
  int mt = rem / 16, nt = rem % 16;
  int w = threadIdx.x >> 5, lane = threadIdx.x & 31;
  int col = lane & 15, hi = lane >> 4;
  int row0 = mt * 128 + w * 16;
  int n0 = nt * 64;

  const bf16*  W    = (mat == 0) ? Wq : ((mat == 1) ? Wk : Wv);
  const float* bias = (mat == 0) ? bq : ((mat == 1) ? bk : bv);

  v8f acc[4];
  #pragma unroll
  for (int c = 0; c < 4; ++c) acc[c] = vzero8();

  // software-pipelined K loop: next-step loads overlap current WMMAs
  v16bf a_cur = load_a_bf16(X, HIDDIM, row0, 0);
  v16bf b_cur[4];
  #pragma unroll
  for (int c = 0; c < 4; ++c) b_cur[c] = load_b_bf16_nk(W, HIDDIM, n0 + c * 16, 0);

  for (int k = 0; k < HIDDIM; k += 32) {
    int kn = k + 32;
    v16bf a_nxt = a_cur;
    v16bf b_nxt[4];
    #pragma unroll
    for (int c = 0; c < 4; ++c) b_nxt[c] = b_cur[c];
    if (kn < HIDDIM) {
      a_nxt = load_a_bf16(X, HIDDIM, row0, kn);
      #pragma unroll
      for (int c = 0; c < 4; ++c)
        b_nxt[c] = load_b_bf16_nk(W, HIDDIM, n0 + c * 16, kn);
      if (kn + 256 < HIDDIM)
        __builtin_prefetch((const void*)(X + (size_t)row0 * HIDDIM + kn + 256), 0, 1);
    }
    #pragma unroll
    for (int c = 0; c < 4; ++c) acc[c] = wmma_bf16(a_cur, b_cur[c], acc[c]);
    a_cur = a_nxt;
    #pragma unroll
    for (int c = 0; c < 4; ++c) b_cur[c] = b_nxt[c];
  }

  #pragma unroll
  for (int c = 0; c < 4; ++c) {
    #pragma unroll
    for (int r = 0; r < 8; ++r) {
      int m = row0 + r + hi * 8;          // global token
      int n = n0 + c * 16 + col;          // feature
      float v = acc[c][r] + bias[n];
      int b = m >> 10, srow = m & (SEQ - 1);
      int head = n >> 6, hd = n & (HDIM - 1);
      if (mat == 0)
        Qb[(((size_t)b * NHEADS + head) * SEQ + srow) * HDIM + hd] = (bf16)v;
      else if (mat == 1)
        Kb[(((size_t)b * NHEADS + head) * SEQ + srow) * HDIM + hd] = (bf16)v;
      else
        Vtb[(((size_t)b * NHEADS + head) * HDIM + hd) * SEQ + srow] = (bf16)v;
    }
  }
}

// ---------------- flash attention + NEZHA relative terms ----------------
// LDS: [0,8K)   K tile double buffer (2 x 32x64 bf16)
//      [8K,16K) V tile double buffer (2 x 64x32 bf16)
//      [16K,..) rel table + per-wave fp32 scratch
__global__ __launch_bounds__(256)
void attn_kernel(const bf16* __restrict__ Qb, const bf16* __restrict__ Kb,
                 const bf16* __restrict__ Vtb, const float* __restrict__ rel,
                 bf16* __restrict__ Ctx) {
  extern __shared__ char smem[];
  // NOTE: no LDS-pointer aggregates (lld can't relocate addrspacecast
  // initializers); buffer pointers are computed from integer offsets.
  bf16*  tab   = (bf16*)(smem + 16384);                       // [TB][HDIM]
  float* fbase = (float*)(smem + 16384 + TB * HDIM * sizeof(bf16));

  int blk = blockIdx.x;
  int rb = blk % (SEQ / 128);
  int hh = (blk / (SEQ / 128)) % NHEADS;
  int b  = blk / ((SEQ / 128) * NHEADS);

  int tid = threadIdx.x;
  // Reconstruct the 129-row sinusoid table from the expanded rel tensor:
  // table[r] = rel[64-r, 0, :] for r<=64, rel[0, r-64, :] for r>=64.
  for (int idx = tid; idx < TB * HDIM; idx += blockDim.x) {
    int r = idx / HDIM, h = idx % HDIM;
    float v = 0.0f;
    if (r <= 64)       v = rel[(size_t)(64 - r) * SEQ * HDIM + h];
    else if (r <= 128) v = rel[(size_t)(r - 64) * HDIM + h];
    tab[idx] = (bf16)v;
  }

  int w = tid >> 5, lane = tid & 31;
  int col = lane & 15, hi = lane >> 4;
  const int WV_FLOATS = 2 * 16 * TB + 16 * 32 + 16;
  float* Pt  = fbase + (size_t)w * WV_FLOATS;  // [16][TB] q.table^T
  float* Pr  = Pt + 16 * TB;                   // [16][TB] prob distance bins
  float* pst = Pr + 16 * TB;                   // [16][32] p-tile staging
  float* rsc = pst + 16 * 32;                  // [16] row rescale factors

  int qrow0 = rb * 128 + w * 16;
  const bf16* qbase = Qb  + ((size_t)(b * NHEADS + hh) * SEQ) * HDIM;
  const bf16* kbase = Kb  + ((size_t)(b * NHEADS + hh) * SEQ) * HDIM;
  const bf16* vbase = Vtb + ((size_t)(b * NHEADS + hh) * HDIM) * SEQ;

  // TDM prologue: wave 0 pulls the first K/V tiles into LDS buffer 0.
  if (tid < 32) {
    tdm_load_2d((const bf16*)smem, kbase, HDIM, SEQ, HDIM, 32, HDIM);
    tdm_load_2d((const bf16*)(smem + 8192), vbase, SEQ, HDIM, 32, HDIM, SEQ);
  }

  __syncthreads();   // rel table ready (TDM completion handled in the loop)

  v16bf qa0 = load_a_bf16(qbase, HDIM, qrow0, 0);
  v16bf qa1 = load_a_bf16(qbase, HDIM, qrow0, 32);

  // Ptile = q_tile @ table^T  (16 x TB), staged to LDS for per-element gather
  for (int rblk = 0; rblk < TB / 16; ++rblk) {
    v16bf b0 = load_b_bf16_nk(tab, HDIM, rblk * 16, 0);
    v16bf b1 = load_b_bf16_nk(tab, HDIM, rblk * 16, 32);
    v8f acc = vzero8();
    acc = wmma_bf16(qa0, b0, acc);
    acc = wmma_bf16(qa1, b1, acc);
    #pragma unroll
    for (int r = 0; r < 8; ++r)
      Pt[(r + hi * 8) * TB + rblk * 16 + col] = acc[r];
  }
  for (int idx = lane; idx < 16 * TB; idx += 32) Pr[idx] = 0.0f;
  asm volatile("s_wait_dscnt 0" ::: "memory");

  float mrow[8], lrow[8];
  #pragma unroll
  for (int r = 0; r < 8; ++r) { mrow[r] = -1e30f; lrow[r] = 0.0f; }
  v8f cacc[4];
  #pragma unroll
  for (int c = 0; c < 4; ++c) cacc[c] = vzero8();

  const float sc = 0.125f;  // 1/sqrt(HDIM)

  for (int jb = 0; jb < SEQ; jb += 32) {
    int cur = (jb >> 5) & 1;
    bf16* Kl = (bf16*)(smem + (cur << 12));            // current K tile
    bf16* Vl = (bf16*)(smem + 8192 + (cur << 12));     // current V tile
    bf16* Kn = (bf16*)(smem + ((cur ^ 1) << 12));      // prefetch K tile
    bf16* Vn = (bf16*)(smem + 8192 + ((cur ^ 1) << 12));
    // wave 0: prefetch next tiles into the other buffer, then guarantee the
    // current buffer's two TDM ops have landed (TDM completes in order, so
    // tensorcnt<=2 leaves only the prefetch outstanding).
    if (tid < 32) {
      if (jb + 32 < SEQ) {
        tdm_load_2d(Kn, kbase + (size_t)(jb + 32) * HDIM,
                    HDIM, SEQ, HDIM, 32, HDIM);
        tdm_load_2d(Vn, vbase + (jb + 32),
                    SEQ, HDIM, 32, HDIM, SEQ);
        __builtin_amdgcn_s_wait_tensorcnt(2);
      } else {
        __builtin_amdgcn_s_wait_tensorcnt(0);
      }
    }
    __syncthreads();  // K/V tiles for this step visible to all waves

    // ---- scores: two 16x16 tiles over 32 keys (B operands from LDS) ----
    v8f s0 = vzero8(), s1 = vzero8();
    {
      v16bf kb0a = load_b_bf16_nk(Kl, HDIM, 0, 0);
      v16bf kb0b = load_b_bf16_nk(Kl, HDIM, 0, 32);
      s0 = wmma_bf16(qa0, kb0a, s0);
      s0 = wmma_bf16(qa1, kb0b, s0);
      v16bf kb1a = load_b_bf16_nk(Kl, HDIM, 16, 0);
      v16bf kb1b = load_b_bf16_nk(Kl, HDIM, 16, 32);
      s1 = wmma_bf16(qa0, kb1a, s1);
      s1 = wmma_bf16(qa1, kb1b, s1);
    }
    // ---- add relative-key scores (Ptile gather) + scale ----
    int drel[2][8];
    #pragma unroll
    for (int jh = 0; jh < 2; ++jh) {
      #pragma unroll
      for (int r = 0; r < 8; ++r) {
        int row = r + hi * 8;
        int i = qrow0 + row;
        int j = jb + jh * 16 + col;
        int d = j - i;
        d = (d < -64) ? -64 : ((d > 64) ? 64 : d);
        drel[jh][r] = row * TB + (d + 64);
        float add = Pt[drel[jh][r]];
        if (jh == 0) s0[r] = (s0[r] + add) * sc;
        else         s1[r] = (s1[r] + add) * sc;
      }
    }
    // ---- online softmax: row reductions across the 16 lanes per half ----
    float alpha[8];
    #pragma unroll
    for (int r = 0; r < 8; ++r) {
      float t = fmaxf(s0[r], s1[r]);
      #pragma unroll
      for (int off = 1; off < 16; off <<= 1)
        t = fmaxf(t, __shfl_xor(t, off, 16));
      float mn = fmaxf(mrow[r], t);
      alpha[r] = __expf(mrow[r] - mn);
      mrow[r] = mn;
      s0[r] = __expf(s0[r] - mn);
      s1[r] = __expf(s1[r] - mn);
      float rs = s0[r] + s1[r];
      #pragma unroll
      for (int off = 1; off < 16; off <<= 1)
        rs += __shfl_xor(rs, off, 16);
      lrow[r] = lrow[r] * alpha[r] + rs;
    }
    // ---- rescale running ctx and distance bins ----
    #pragma unroll
    for (int c = 0; c < 4; ++c)
      #pragma unroll
      for (int r = 0; r < 8; ++r) cacc[c][r] *= alpha[r];
    #pragma unroll
    for (int r = 0; r < 8; ++r)
      if (col == r) rsc[hi * 8 + r] = alpha[r];
    asm volatile("s_wait_dscnt 0" ::: "memory");
    for (int idx = lane; idx < 16 * TB; idx += 32)
      Pr[idx] *= rsc[idx / TB];
    // ---- scatter probs into distance bins + stage for PV transpose ----
    #pragma unroll
    for (int jh = 0; jh < 2; ++jh) {
      #pragma unroll
      for (int r = 0; r < 8; ++r) {
        int row = r + hi * 8;
        float p = jh ? s1[r] : s0[r];
        atomicAdd(&Pr[drel[jh][r]], p);
        pst[row * 32 + jh * 16 + col] = p;
      }
    }
    asm volatile("s_wait_dscnt 0" ::: "memory");
    // ---- read p back as A-operand (16x32, C-layout -> A-layout via LDS) ----
    v16bf pa;
    #pragma unroll
    for (int t = 0; t < 8; ++t) {
      pa[t]     = (bf16)pst[col * 32 + hi * 8 + t];
      pa[8 + t] = (bf16)pst[col * 32 + 16 + hi * 8 + t];
    }
    // ---- ctx += p @ V  (V tile in LDS, transposed: contiguous key strips) ----
    #pragma unroll
    for (int c = 0; c < 4; ++c) {
      v16bf vb = load_b_bf16_nk(Vl, 32, c * 16, 0);
      cacc[c] = wmma_bf16(pa, vb, cacc[c]);
    }
    __syncthreads();  // all waves done reading before buffer is re-prefetched
  }

  asm volatile("s_wait_dscnt 0" ::: "memory");
  // ---- relative-value term: ctx += bins @ table ----
  for (int kc = 0; kc < TB / 32; ++kc) {
    v16bf pa;
    #pragma unroll
    for (int t = 0; t < 8; ++t) {
      pa[t]     = (bf16)Pr[col * TB + kc * 32 + hi * 8 + t];
      pa[8 + t] = (bf16)Pr[col * TB + kc * 32 + 16 + hi * 8 + t];
    }
    #pragma unroll
    for (int c = 0; c < 4; ++c) {
      v16bf tb2 = load_b_bf16_kn(tab, HDIM, kc * 32, c * 16);
      cacc[c] = wmma_bf16(pa, tb2, cacc[c]);
    }
  }

  // ---- normalize and write ctx as [B, S, NH*HD] bf16 ----
  #pragma unroll
  for (int r = 0; r < 8; ++r) {
    float inv = 1.0f / lrow[r];
    int row = r + hi * 8;
    size_t tok = (size_t)b * SEQ + qrow0 + row;
    #pragma unroll
    for (int c = 0; c < 4; ++c) {
      float v = cacc[c][r] * inv;
      Ctx[tok * HIDDIM + hh * HDIM + c * 16 + col] = (bf16)v;
    }
  }
}

// ---------------- output projection: out = ctx @ Wo^T + bo (fp32 out) ----------------
__global__ __launch_bounds__(256)
void oproj_gemm_kernel(const bf16* __restrict__ A, const bf16* __restrict__ W,
                       const float* __restrict__ bias, float* __restrict__ out) {
  int blk = blockIdx.x;
  int mt = blk / 16, nt = blk % 16;
  int w = threadIdx.x >> 5, lane = threadIdx.x & 31;
  int col = lane & 15, hi = lane >> 4;
  int row0 = mt * 128 + w * 16;
  int n0 = nt * 64;

  v8f acc[4];
  #pragma unroll
  for (int c = 0; c < 4; ++c) acc[c] = vzero8();

  v16bf a_cur = load_a_bf16(A, HIDDIM, row0, 0);
  v16bf b_cur[4];
  #pragma unroll
  for (int c = 0; c < 4; ++c) b_cur[c] = load_b_bf16_nk(W, HIDDIM, n0 + c * 16, 0);

  for (int k = 0; k < HIDDIM; k += 32) {
    int kn = k + 32;
    v16bf a_nxt = a_cur;
    v16bf b_nxt[4];
    #pragma unroll
    for (int c = 0; c < 4; ++c) b_nxt[c] = b_cur[c];
    if (kn < HIDDIM) {
      a_nxt = load_a_bf16(A, HIDDIM, row0, kn);
      #pragma unroll
      for (int c = 0; c < 4; ++c)
        b_nxt[c] = load_b_bf16_nk(W, HIDDIM, n0 + c * 16, kn);
      if (kn + 256 < HIDDIM)
        __builtin_prefetch((const void*)(A + (size_t)row0 * HIDDIM + kn + 256), 0, 1);
    }
    #pragma unroll
    for (int c = 0; c < 4; ++c) acc[c] = wmma_bf16(a_cur, b_cur[c], acc[c]);
    a_cur = a_nxt;
    #pragma unroll
    for (int c = 0; c < 4; ++c) b_cur[c] = b_nxt[c];
  }

  #pragma unroll
  for (int c = 0; c < 4; ++c)
    #pragma unroll
    for (int r = 0; r < 8; ++r) {
      int m = row0 + r + hi * 8;
      int n = n0 + c * 16 + col;
      out[(size_t)m * HIDDIM + n] = acc[c][r] + bias[n];
    }
}

// ---------------- host launcher ----------------
extern "C" void kernel_launch(void* const* d_in, const int* in_sizes, int n_in,
                              void* d_out, int out_size, void* d_ws, size_t ws_size,
                              hipStream_t stream) {
  const float* hs  = (const float*)d_in[0];
  const float* Wq  = (const float*)d_in[1];
  const float* bq  = (const float*)d_in[2];
  const float* Wk  = (const float*)d_in[3];
  const float* bk  = (const float*)d_in[4];
  const float* Wv  = (const float*)d_in[5];
  const float* bv  = (const float*)d_in[6];
  const float* Wo  = (const float*)d_in[7];
  const float* bo  = (const float*)d_in[8];
  const float* rel = (const float*)d_in[9];
  float* out = (float*)d_out;

  char* ws = (char*)d_ws;
  const size_t MB = 1024ull * 1024ull;
  bf16* Xb  = (bf16*)(ws + 0 * MB);
  bf16* Wqb = (bf16*)(ws + 16 * MB);
  bf16* Wkb = (bf16*)(ws + 18 * MB);
  bf16* Wvb = (bf16*)(ws + 20 * MB);
  bf16* Wob = (bf16*)(ws + 22 * MB);
  bf16* Qb  = (bf16*)(ws + 24 * MB);
  bf16* Kb  = (bf16*)(ws + 40 * MB);
  bf16* Vtb = (bf16*)(ws + 56 * MB);
  bf16* Ctx = (bf16*)(ws + 72 * MB);

  const int nX = BATCH * SEQ * HIDDIM;      // 8M
  const int nW = HIDDIM * HIDDIM;           // 1M
  f32_to_bf16_kernel<<<(nX + 255) / 256, 256, 0, stream>>>(hs, Xb, nX);
  f32_to_bf16_kernel<<<(nW + 255) / 256, 256, 0, stream>>>(Wq, Wqb, nW);
  f32_to_bf16_kernel<<<(nW + 255) / 256, 256, 0, stream>>>(Wk, Wkb, nW);
  f32_to_bf16_kernel<<<(nW + 255) / 256, 256, 0, stream>>>(Wv, Wvb, nW);
  f32_to_bf16_kernel<<<(nW + 255) / 256, 256, 0, stream>>>(Wo, Wob, nW);

  qkv_gemm_kernel<<<3 * 64 * 16, 256, 0, stream>>>(Xb, Wqb, Wkb, Wvb,
                                                   bq, bk, bv, Qb, Kb, Vtb);

  size_t shmem = 16384 +                                   // K/V TDM buffers
                 (size_t)TB * HDIM * sizeof(bf16) +        // rel table
                 8ull * (2 * 16 * TB + 16 * 32 + 16) * sizeof(float);
  attn_kernel<<<BATCH * NHEADS * (SEQ / 128), 256, shmem, stream>>>(
      Qb, Kb, Vtb, rel, Ctx);

  oproj_gemm_kernel<<<64 * 16, 256, 0, stream>>>(Ctx, Wob, bo, out);
}
